// GAT_MS_33251636806227
// MI455X (gfx1250) — matrix operand order repaired and checked by reading, not compiled, not run on hardware
//
#include <hip/hip_runtime.h>
#include <cmath>

#define NN    4096
#define NE    131072
#define FEATD 64
#define HIDD  256
#define NHEAD 4
#define DHH   64

typedef __attribute__((ext_vector_type(16))) __bf16         v16bf;
typedef __attribute__((ext_vector_type(8)))  float          v8f;
typedef __attribute__((ext_vector_type(8)))  unsigned short u16x8;
typedef __attribute__((ext_vector_type(4)))  int            v4i;

#define AS1 __attribute__((address_space(1)))
#define AS3 __attribute__((address_space(3)))

#if defined(__has_builtin)
#if __has_builtin(__builtin_amdgcn_global_load_async_to_lds_b128) && \
    __has_builtin(__builtin_amdgcn_s_wait_asynccnt)
#define USE_ASYNC_LDS 1
#endif
#endif
#ifndef USE_ASYNC_LDS
#define USE_ASYNC_LDS 0
#endif

union BFrag {
  struct { u16x8 lo, hi; } p;
  v16bf v;
};

__device__ __forceinline__ unsigned short f2bf(float f) {
  unsigned int u = __builtin_bit_cast(unsigned int, f);
  unsigned int r = (u + 0x7FFFu + ((u >> 16) & 1u)) >> 16;   // round-to-nearest-even
  return (unsigned short)r;
}

// ---------------------------------------------------------------------------
// bf16-WMMA GEMM: C[4096,256] = act(concat(A1[:,:K1],A2)[4096,K] @ W[K,256] + bias)
// A operands are pre-converted bf16 row-major; weights pre-transposed bf16
// Bt[256][K]. Tiles staged to LDS with GLOBAL_LOAD_ASYNC_TO_LDS_B128 (ASYNCcnt).
// Block: 128 threads = 4 waves, each wave a 16x64 tile; WG tile 64x64.
// ---------------------------------------------------------------------------
__global__ __launch_bounds__(128)
void gemm_bf16_kernel(const unsigned short* __restrict__ A1,
                      const unsigned short* __restrict__ A2,
                      int K1, int K, const unsigned short* __restrict__ Bt,
                      const float* __restrict__ bias, int relu,
                      float* __restrict__ Cf, unsigned short* __restrict__ Cb, int Ncol) {
  __shared__ unsigned short lA[64][32];    // [row][k]
  __shared__ unsigned short lBt[64][32];   // [col][k]
  const int tid = threadIdx.x;
  const int wave = tid >> 5, lane = tid & 31;
  const int n16 = lane & 15, kh = lane >> 4;
  const int bm = blockIdx.x * 64;
  const int bn = blockIdx.y * 64;
  const int wrow = bm + wave * 16;

  v8f acc[4] = {};

  for (int k0 = 0; k0 < K; k0 += 32) {
    __syncthreads();                       // previous tile fully consumed
#if USE_ASYNC_LDS
    for (int t = tid; t < 256; t += 128) {
      const int r = t >> 2, seg = t & 3;   // 4 x 16B segments per 64B row
      const int gr = bm + r;
      const unsigned short* ga = (k0 < K1)
          ? (A1 + (size_t)gr * K1 + k0)
          : (A2 + (size_t)gr * (K - K1) + (k0 - K1));
      __builtin_amdgcn_global_load_async_to_lds_b128(
          (AS1 v4i*)(ga + seg * 8), (AS3 v4i*)&lA[r][seg * 8], 0, 0);
      const unsigned short* gb = Bt + (size_t)(bn + r) * K + k0;
      __builtin_amdgcn_global_load_async_to_lds_b128(
          (AS1 v4i*)(gb + seg * 8), (AS3 v4i*)&lBt[r][seg * 8], 0, 0);
    }
    __builtin_amdgcn_s_wait_asynccnt(0);
    __syncthreads();
#else
    for (int i = tid; i < 64 * 32; i += 128) {
      const int r = i >> 5, c = i & 31;
      const int gr = bm + r;
      const unsigned short* ga = (k0 < K1)
          ? (A1 + (size_t)gr * K1 + k0)
          : (A2 + (size_t)gr * (K - K1) + (k0 - K1));
      lA[r][c]  = ga[c];
      lBt[r][c] = Bt[(size_t)(bn + r) * K + k0 + c];
    }
    __syncthreads();
#endif

    BFrag af;
    const int ar = wave * 16 + n16;
    af.p.lo = *(const u16x8*)&lA[ar][kh * 8];
    af.p.hi = *(const u16x8*)&lA[ar][16 + kh * 8];
#pragma unroll
    for (int c = 0; c < 4; ++c) {
      BFrag bf_;
      const int bc = c * 16 + n16;
      bf_.p.lo = *(const u16x8*)&lBt[bc][kh * 8];
      bf_.p.hi = *(const u16x8*)&lBt[bc][16 + kh * 8];
      acc[c] = __builtin_amdgcn_wmma_f32_16x16x32_bf16(false, af.v, false, bf_.v,
                                                       (short)0, acc[c], false, false);
    }
  }

  // epilogue: C layout -> VGPR r / lane n16 / half kh
#pragma unroll
  for (int c = 0; c < 4; ++c) {
#pragma unroll
    for (int r = 0; r < 8; ++r) {
      const int row = wrow + kh * 8 + r;
      const int col = bn + c * 16 + n16;
      float v = acc[c][r];
      if (bias) v += bias[col];
      if (relu) v = v > 0.f ? v : 0.f;
      if (Cf) Cf[(size_t)row * Ncol + col] = v;
      if (Cb) Cb[(size_t)row * Ncol + col] = f2bf(v);
    }
  }
}

// ---------------------------------------------------------------------------
// Mean-shift banded attention (flash-style): one wave per 16 query rows.
// scores = (QK^T)/16 - |y_i-y_j|, masked by distance_mask & dist2<=100, online
// softmax, out = attn@V (WMMA, bf16 out) and attn@xyz (VALU).
// ---------------------------------------------------------------------------
__global__ __launch_bounds__(128)
void ms_attn_kernel(const unsigned short* __restrict__ Qb, const unsigned short* __restrict__ Kb,
                    const unsigned short* __restrict__ Vt, const float* __restrict__ yhat,
                    const unsigned char* __restrict__ dmask, const unsigned char* __restrict__ bmask,
                    const float* __restrict__ xyz_in, unsigned short* __restrict__ hmsb,
                    float* __restrict__ xyz_out) {
  __shared__ unsigned short lP[4][16][32];   // per-wave P transpose staging
  const int tid = threadIdx.x;
  const int wave = tid >> 5, lane = tid & 31;
  const int n16 = lane & 15, kh = lane >> 4;
  const int i0 = (blockIdx.x * 4 + wave) * 16;

  // preload Q A-frags (16x32 per chunk, 8 chunks over K=256)
  v16bf qf[8];
  {
    const unsigned short* qrow = Qb + (size_t)(i0 + n16) * HIDD;
#pragma unroll
    for (int c = 0; c < 8; ++c) {
      BFrag u;
      u.p.lo = *(const u16x8*)(qrow + c * 32 + kh * 8);
      u.p.hi = *(const u16x8*)(qrow + c * 32 + 16 + kh * 8);
      qf[c] = u.v;
    }
  }

  float yr[8], xr0[8], xr1[8], xr2[8], mrow[8], lrow[8], ax[8], ay[8], az[8];
#pragma unroll
  for (int r = 0; r < 8; ++r) {
    const int row = i0 + kh * 8 + r;
    yr[r]  = yhat[row];
    xr0[r] = xyz_in[row * 3 + 0];
    xr1[r] = xyz_in[row * 3 + 1];
    xr2[r] = xyz_in[row * 3 + 2];
    mrow[r] = -3.0e38f; lrow[r] = 0.f;
    ax[r] = 0.f; ay[r] = 0.f; az[r] = 0.f;
  }
  v8f acc[16] = {};
  const float invscale = 1.0f / 16.0f;   // 1/(sqrt(HID)*T)

  for (int j0 = 0; j0 < NN; j0 += 32) {
    if (j0 + 32 < NN)
      __builtin_prefetch(Kb + (size_t)(j0 + 32 + n16) * HIDD, 0, 1);  // global_prefetch_b8

    float yc[2], xc0[2], xc1[2], xc2[2];
#pragma unroll
    for (int t = 0; t < 2; ++t) {
      const int col = j0 + t * 16 + n16;
      yc[t]  = yhat[col];
      xc0[t] = xyz_in[col * 3 + 0];
      xc1[t] = xyz_in[col * 3 + 1];
      xc2[t] = xyz_in[col * 3 + 2];
    }

    // S = Q @ K^T for two 16-key subtiles
    v8f S[2] = {};
#pragma unroll
    for (int c = 0; c < 8; ++c) {
#pragma unroll
      for (int t = 0; t < 2; ++t) {
        BFrag kf;
        const unsigned short* kr = Kb + (size_t)(j0 + t * 16 + n16) * HIDD + c * 32 + kh * 8;
        kf.p.lo = *(const u16x8*)(kr);
        kf.p.hi = *(const u16x8*)(kr + 16);
        S[t] = __builtin_amdgcn_wmma_f32_16x16x32_bf16(false, qf[c], false, kf.v,
                                                       (short)0, S[t], false, false);
      }
    }

    // bias + band mask
#pragma unroll
    for (int t = 0; t < 2; ++t) {
      const int col = j0 + t * 16 + n16;
#pragma unroll
      for (int r = 0; r < 8; ++r) {
        const int row = i0 + kh * 8 + r;
        const float s = S[t][r] * invscale - fabsf(yr[r] - yc[t]);   // BETA = 1
        const float dx = xr0[r] - xc0[t], dy = xr1[r] - xc1[t], dz = xr2[r] - xc2[t];
        const float d2 = dx * dx + dy * dy + dz * dz;
        const bool ok = dmask[(size_t)row * NN + col] && bmask[(size_t)row * NN + col]
                        && (d2 <= 100.0f);
        S[t][r] = ok ? s : -1.0e9f;
      }
    }

    // online softmax (row reductions across the 16 lanes of each half-wave)
    float mnew[8], scal[8];
#pragma unroll
    for (int r = 0; r < 8; ++r) {
      float mt = fmaxf(S[0][r], S[1][r]);
#pragma unroll
      for (int off = 1; off < 16; off <<= 1)
        mt = fmaxf(mt, __shfl_xor(mt, off, 16));
      mnew[r] = fmaxf(mrow[r], mt);
      scal[r] = __expf(mrow[r] - mnew[r]);
      mrow[r] = mnew[r];
    }

#pragma unroll
    for (int r = 0; r < 8; ++r) {
      const float p0 = __expf(S[0][r] - mnew[r]);
      const float p1 = __expf(S[1][r] - mnew[r]);
      lrow[r] = lrow[r] * scal[r] + p0 + p1;
      ax[r] = ax[r] * scal[r] + p0 * xc0[0] + p1 * xc0[1];
      ay[r] = ay[r] * scal[r] + p0 * xc1[0] + p1 * xc1[1];
      az[r] = az[r] * scal[r] + p0 * xc2[0] + p1 * xc2[1];
      lP[wave][kh * 8 + r][n16]      = f2bf(p0);
      lP[wave][kh * 8 + r][16 + n16] = f2bf(p1);
    }

    // rescale running accumulators
#pragma unroll
    for (int c = 0; c < 16; ++c)
#pragma unroll
      for (int r = 0; r < 8; ++r)
        acc[c][r] *= scal[r];

    __builtin_amdgcn_wave_barrier();   // keep LDS store->load order for the transpose

    // P in A-matrix layout from LDS
    BFrag pf;
    pf.p.lo = *(const u16x8*)&lP[wave][n16][kh * 8];
    pf.p.hi = *(const u16x8*)&lP[wave][n16][16 + kh * 8];

    // acc += P @ V  (V pre-transposed: Vt[feature][key])
#pragma unroll
    for (int c = 0; c < 16; ++c) {
      BFrag vf;
      const unsigned short* vr = Vt + (size_t)(c * 16 + n16) * NN + j0 + kh * 8;
      vf.p.lo = *(const u16x8*)(vr);
      vf.p.hi = *(const u16x8*)(vr + 16);
      acc[c] = __builtin_amdgcn_wmma_f32_16x16x32_bf16(false, pf.v, false, vf.v,
                                                       (short)0, acc[c], false, false);
    }
  }

  // finalize: normalize, store bf16 hms + new xyz
#pragma unroll
  for (int r = 0; r < 8; ++r) {
    float lt = lrow[r];
#pragma unroll
    for (int off = 1; off < 16; off <<= 1) lt += __shfl_xor(lt, off, 16);
    const float inv = 1.0f / (lt + 1e-30f);
    const int row = i0 + kh * 8 + r;
#pragma unroll
    for (int c = 0; c < 16; ++c)
      hmsb[(size_t)row * HIDD + c * 16 + n16] = f2bf(acc[c][r] * inv);
    float a0 = ax[r], a1 = ay[r], a2 = az[r];
#pragma unroll
    for (int off = 1; off < 16; off <<= 1) {
      a0 += __shfl_xor(a0, off, 16);
      a1 += __shfl_xor(a1, off, 16);
      a2 += __shfl_xor(a2, off, 16);
    }
    if (n16 == 0) {
      xyz_out[row * 3 + 0] = a0 * inv;
      xyz_out[row * 3 + 1] = a1 * inv;
      xyz_out[row * 3 + 2] = a2 * inv;
    }
  }
}

// ---------------------------------------------------------------------------
// prep / conversion kernels
// ---------------------------------------------------------------------------
__global__ void prep_wt_kernel(const float* __restrict__ W, unsigned short* __restrict__ Wt,
                               int K) {   // W[K][256] f32 -> Wt[256][K] bf16
  const int idx = blockIdx.x * blockDim.x + threadIdx.x;
  if (idx >= K * HIDD) return;
  const int k = idx >> 8, c = idx & 255;
  Wt[(size_t)c * K + k] = f2bf(W[idx]);
}

__global__ void cvt_bf16_kernel(const float* __restrict__ s, unsigned short* __restrict__ d,
                                int n) {
  const int idx = blockIdx.x * blockDim.x + threadIdx.x;
  if (idx < n) d[idx] = f2bf(s[idx]);
}

__global__ void transpose_kernel(const unsigned short* __restrict__ vb,
                                 unsigned short* __restrict__ vt) {
  const int idx = blockIdx.x * blockDim.x + threadIdx.x;
  const int n = idx >> 8, f = idx & 255;
  vt[(size_t)f * NN + n] = vb[(size_t)n * HIDD + f];
}

// ---------------------------------------------------------------------------
// GAT helpers
// ---------------------------------------------------------------------------
__global__ void gat_elr_kernel(const float* __restrict__ h, const float* __restrict__ al,
                               const float* __restrict__ ar, float* __restrict__ el,
                               float* __restrict__ er) {
  const int wave = threadIdx.x >> 5, lane = threadIdx.x & 31;
  const int idx = blockIdx.x * 8 + wave;          // node*NHEAD + head
  const int node = idx >> 2, head = idx & 3;
  const float* hp  = h + (size_t)node * HIDD + head * DHH;
  const float* alp = al + head * DHH;
  const float* arp = ar + head * DHH;
  float sl = 0.f, sr = 0.f;
#pragma unroll
  for (int d = lane; d < DHH; d += 32) { const float v = hp[d]; sl += v * alp[d]; sr += v * arp[d]; }
#pragma unroll
  for (int off = 1; off < 32; off <<= 1) { sl += __shfl_xor(sl, off, 32); sr += __shfl_xor(sr, off, 32); }
  if (lane == 0) { el[idx] = sl; er[idx] = sr; }
}

__global__ void gat_init_kernel(float* xg, float* mb, float* zb) {
  const int idx = blockIdx.x * blockDim.x + threadIdx.x;
  if (idx < NN * HIDD) xg[idx] = 0.f;
  if (idx < NN * NHEAD) { mb[idx] = -3.0e38f; zb[idx] = 0.f; }
}

__device__ __forceinline__ void atomicMaxF(float* addr, float v) {
  if (v >= 0.f) atomicMax((int*)addr, __float_as_int(v));
  else          atomicMin((unsigned int*)addr, __float_as_uint(v));
}

__global__ void gat_edge_max_kernel(const int* __restrict__ src, const int* __restrict__ dst,
                                    const float* __restrict__ el, const float* __restrict__ er,
                                    float* __restrict__ ebuf, float* __restrict__ mb) {
  const int t = blockIdx.x * blockDim.x + threadIdx.x;
  if (t >= NE * NHEAD) return;
  const int e = t >> 2, hh = t & 3;
  const int s = src[e], d = dst[e];
  float x = el[s * NHEAD + hh] + er[d * NHEAD + hh];
  x = x > 0.f ? x : 0.2f * x;                       // leaky_relu(0.2)
  ebuf[t] = x;
  atomicMaxF(&mb[d * NHEAD + hh], x);
}

__global__ void gat_edge_sum_kernel(const int* __restrict__ dst, const float* __restrict__ ebuf,
                                    const float* __restrict__ mb, float* __restrict__ wbuf,
                                    float* __restrict__ zb) {
  const int t = blockIdx.x * blockDim.x + threadIdx.x;
  if (t >= NE * NHEAD) return;
  const int e = t >> 2, hh = t & 3;
  const int d = dst[e];
  const float w = __expf(ebuf[t] - mb[d * NHEAD + hh]);
  wbuf[t] = w;
  atomicAdd(&zb[d * NHEAD + hh], w);
}

__global__ void gat_aggr_kernel(const int* __restrict__ src, const int* __restrict__ dst,
                                const float* __restrict__ wbuf, const float* __restrict__ zb,
                                const float* __restrict__ h, float* __restrict__ xg) {
  const int e = blockIdx.x, f = threadIdx.x, hh = f >> 6;
  const int s = src[e], d = dst[e];
  const float coef = wbuf[e * NHEAD + hh] / (zb[d * NHEAD + hh] + 1e-9f);
  atomicAdd(&xg[(size_t)d * HIDD + f], coef * h[(size_t)s * HIDD + f]);
}

__global__ void gcnii_kernel(float* __restrict__ x, const float* __restrict__ tmp,
                             const float* __restrict__ xg, const float* __restrict__ h0,
                             float theta, unsigned short* __restrict__ xb) {
  const int idx = blockIdx.x * blockDim.x + threadIdx.x;
  const float v = theta * tmp[idx] + (1.f - theta) * (0.9f * xg[idx] + 0.1f * h0[idx]) + x[idx];
  x[idx] = v;
  xb[idx] = f2bf(v);
}

__global__ void add_kernel(float* __restrict__ x, const float* __restrict__ t,
                           unsigned short* __restrict__ xb) {
  const int idx = blockIdx.x * blockDim.x + threadIdx.x;
  const float v = x[idx] + t[idx];
  x[idx] = v;
  xb[idx] = f2bf(v);
}

// tiny 256->2 heads: one wave per node
__global__ void head2_kernel(const float* __restrict__ x, const float* __restrict__ W,
                             const float* __restrict__ b, float* __restrict__ out,
                             int yhat_mode) {
  const int wave = threadIdx.x >> 5, lane = threadIdx.x & 31;
  const int node = blockIdx.x * 8 + wave;
  const float* xp = x + (size_t)node * HIDD;
  float s0 = 0.f, s1 = 0.f;
#pragma unroll
  for (int d = lane; d < HIDD; d += 32) { const float v = xp[d]; s0 += v * W[d * 2]; s1 += v * W[d * 2 + 1]; }
#pragma unroll
  for (int off = 1; off < 32; off <<= 1) { s0 += __shfl_xor(s0, off, 32); s1 += __shfl_xor(s1, off, 32); }
  if (lane == 0) {
    s0 += b[0]; s1 += b[1];
    if (yhat_mode) {
      const float m = fmaxf(s0, s1);
      const float e0 = __expf(s0 - m), e1 = __expf(s1 - m);
      out[node] = e1 / (e0 + e1);
    } else {
      out[node * 2 + 0] = s0;
      out[node * 2 + 1] = s1;
    }
  }
}

// ---------------------------------------------------------------------------
extern "C" void kernel_launch(void* const* d_in, const int* in_sizes, int n_in,
                              void* d_out, int out_size, void* d_ws, size_t ws_size,
                              hipStream_t stream) {
  (void)in_sizes; (void)n_in; (void)out_size; (void)ws_size;
  const float* feat  = (const float*)d_in[0];
  const float* xyz   = (const float*)d_in[1];
  const int*   srcI  = (const int*)d_in[2];
  const int*   dstI  = (const int*)d_in[3];
  const unsigned char* dmask = (const unsigned char*)d_in[4];
  const unsigned char* bmask = (const unsigned char*)d_in[5];
  const float* fc_W = (const float*)d_in[6];
  const float* fc_b = (const float*)d_in[7];
  const float* gat_W = (const float*)d_in[8];
  const float* attn_l = (const float*)d_in[9];
  const float* attn_r = (const float*)d_in[10];
  const float* gcnii_W = (const float*)d_in[11];
  const float* cls_gat_W = (const float*)d_in[12];
  const float* cls_gat_b = (const float*)d_in[13];
  const float* q_W = (const float*)d_in[14];
  const float* q_b = (const float*)d_in[15];
  const float* k_W = (const float*)d_in[16];
  const float* k_b = (const float*)d_in[17];
  const float* v_W = (const float*)d_in[18];
  const float* v_b = (const float*)d_in[19];
  const float* o_W = (const float*)d_in[20];
  const float* o_b = (const float*)d_in[21];
  const float* cls_W = (const float*)d_in[22];
  const float* cls_b = (const float*)d_in[23];

  char* wptr = (char*)d_ws;
  auto alloc = [&](size_t bytes) { char* p = wptr; wptr += (bytes + 255) & ~(size_t)255; return p; };
  // f32 activations
  float* x   = (float*)alloc((size_t)NN * HIDD * 4);
  float* h0  = (float*)alloc((size_t)NN * HIDD * 4);
  float* hb  = (float*)alloc((size_t)NN * HIDD * 4);
  float* xg  = (float*)alloc((size_t)NN * HIDD * 4);
  float* tmp = (float*)alloc((size_t)NN * HIDD * 4);
  // bf16 mirrors
  unsigned short* featb = (unsigned short*)alloc((size_t)NN * FEATD * 2);
  unsigned short* xb    = (unsigned short*)alloc((size_t)NN * HIDD * 2);
  unsigned short* h0b   = (unsigned short*)alloc((size_t)NN * HIDD * 2);
  unsigned short* xgb   = (unsigned short*)alloc((size_t)NN * HIDD * 2);
  unsigned short* hmsb  = (unsigned short*)alloc((size_t)NN * HIDD * 2);
  unsigned short* qb = (unsigned short*)alloc((size_t)NN * HIDD * 2);
  unsigned short* kb = (unsigned short*)alloc((size_t)NN * HIDD * 2);
  unsigned short* vb = (unsigned short*)alloc((size_t)NN * HIDD * 2);
  unsigned short* vt = (unsigned short*)alloc((size_t)NN * HIDD * 2);
  // bf16 transposed weights
  unsigned short* fcWt    = (unsigned short*)alloc((size_t)FEATD * HIDD * 2);
  unsigned short* gatWt   = (unsigned short*)alloc((size_t)4 * HIDD * HIDD * 2);
  unsigned short* gcniiWt = (unsigned short*)alloc((size_t)4 * 2 * HIDD * HIDD * 2);
  unsigned short* qWt = (unsigned short*)alloc((size_t)4 * HIDD * HIDD * 2);
  unsigned short* kWt = (unsigned short*)alloc((size_t)4 * HIDD * HIDD * 2);
  unsigned short* vWt = (unsigned short*)alloc((size_t)4 * HIDD * HIDD * 2);
  unsigned short* oWt = (unsigned short*)alloc((size_t)4 * HIDD * HIDD * 2);
  // GAT edge scratch
  float* el   = (float*)alloc((size_t)NN * NHEAD * 4);
  float* er   = (float*)alloc((size_t)NN * NHEAD * 4);
  float* mb   = (float*)alloc((size_t)NN * NHEAD * 4);
  float* zb   = (float*)alloc((size_t)NN * NHEAD * 4);
  float* ebuf = (float*)alloc((size_t)NE * NHEAD * 4);
  float* wbuf = (float*)alloc((size_t)NE * NHEAD * 4);
  float* yhat = (float*)alloc((size_t)NN * 4);
  float* xyzA = (float*)alloc((size_t)NN * 3 * 4);
  float* xyzB = (float*)alloc((size_t)NN * 3 * 4);

  // ---- weight prep (f32 -> bf16, transposed to [256][K]) ----
  prep_wt_kernel<<<FEATD, 256, 0, stream>>>(fc_W, fcWt, FEATD);
  for (int l = 0; l < 4; ++l) {
    prep_wt_kernel<<<HIDD, 256, 0, stream>>>(gat_W + (size_t)l * HIDD * HIDD,
                                             gatWt + (size_t)l * HIDD * HIDD, HIDD);
    prep_wt_kernel<<<2 * HIDD, 256, 0, stream>>>(gcnii_W + (size_t)l * 2 * HIDD * HIDD,
                                                 gcniiWt + (size_t)l * 2 * HIDD * HIDD, 2 * HIDD);
    prep_wt_kernel<<<HIDD, 256, 0, stream>>>(q_W + (size_t)l * HIDD * HIDD,
                                             qWt + (size_t)l * HIDD * HIDD, HIDD);
    prep_wt_kernel<<<HIDD, 256, 0, stream>>>(k_W + (size_t)l * HIDD * HIDD,
                                             kWt + (size_t)l * HIDD * HIDD, HIDD);
    prep_wt_kernel<<<HIDD, 256, 0, stream>>>(v_W + (size_t)l * HIDD * HIDD,
                                             vWt + (size_t)l * HIDD * HIDD, HIDD);
    prep_wt_kernel<<<HIDD, 256, 0, stream>>>(o_W + (size_t)l * HIDD * HIDD,
                                             oWt + (size_t)l * HIDD * HIDD, HIDD);
  }
  cvt_bf16_kernel<<<NN * FEATD / 256, 256, 0, stream>>>(feat, featb, NN * FEATD);

  const dim3 gGemm(NN / 64, HIDD / 64);
  auto gemm = [&](const unsigned short* A1, const unsigned short* A2, int K1, int K,
                  const unsigned short* Bt, const float* bias, int relu,
                  float* Cf, unsigned short* Cb) {
    gemm_bf16_kernel<<<gGemm, 128, 0, stream>>>(A1, A2, K1, K, Bt, bias, relu, Cf, Cb, HIDD);
  };

  // h0 = x = relu(feat @ fc_W + fc_b)
  gemm(featb, featb, FEATD, FEATD, fcWt, fc_b, 1, x, xb);
  (void)hipMemcpyAsync(h0,  x,  (size_t)NN * HIDD * 4, hipMemcpyDeviceToDevice, stream);
  (void)hipMemcpyAsync(h0b, xb, (size_t)NN * HIDD * 2, hipMemcpyDeviceToDevice, stream);

  // GAT + GCNII stack
  for (int l = 0; l < 4; ++l) {
    gemm(xb, xb, HIDD, HIDD, gatWt + (size_t)l * HIDD * HIDD, nullptr, 0, hb, nullptr);
    gat_elr_kernel<<<NN * NHEAD / 8, 256, 0, stream>>>(hb, attn_l + l * NHEAD * DHH,
                                                       attn_r + l * NHEAD * DHH, el, er);
    gat_init_kernel<<<NN * HIDD / 256, 256, 0, stream>>>(xg, mb, zb);
    gat_edge_max_kernel<<<NE * NHEAD / 256, 256, 0, stream>>>(srcI, dstI, el, er, ebuf, mb);
    gat_edge_sum_kernel<<<NE * NHEAD / 256, 256, 0, stream>>>(dstI, ebuf, mb, wbuf, zb);
    gat_aggr_kernel<<<NE, 256, 0, stream>>>(srcI, dstI, wbuf, zb, hb, xg);
    cvt_bf16_kernel<<<NN * HIDD / 256, 256, 0, stream>>>(xg, xgb, NN * HIDD);
    gemm(xgb, h0b, HIDD, 2 * HIDD, gcniiWt + (size_t)l * 2 * HIDD * HIDD, nullptr, 0, tmp, nullptr);
    const float theta = fminf(1.0f, logf(0.5f / (float)(l + 1) + 1.0f));
    gcnii_kernel<<<NN * HIDD / 256, 256, 0, stream>>>(x, tmp, xg, h0, theta, xb);
  }

  // y_hat = softmax(x @ cls_gat_W + b)[:,1]
  head2_kernel<<<NN / 8, 256, 0, stream>>>(x, cls_gat_W, cls_gat_b, yhat, 1);
  (void)hipMemcpyAsync(xyzA, xyz, (size_t)NN * 3 * 4, hipMemcpyDeviceToDevice, stream);

  // mean-shift stack
  float* xa = xyzA; float* xbz = xyzB;
  for (int l = 0; l < 4; ++l) {
    gemm(xb, xb, HIDD, HIDD, qWt + (size_t)l * HIDD * HIDD, q_b + l * HIDD, 0, nullptr, qb);
    gemm(xb, xb, HIDD, HIDD, kWt + (size_t)l * HIDD * HIDD, k_b + l * HIDD, 0, nullptr, kb);
    gemm(xb, xb, HIDD, HIDD, vWt + (size_t)l * HIDD * HIDD, v_b + l * HIDD, 0, nullptr, vb);
    transpose_kernel<<<NN * HIDD / 256, 256, 0, stream>>>(vb, vt);
    ms_attn_kernel<<<NN / 64, 128, 0, stream>>>(qb, kb, vt, yhat, dmask, bmask, xa, hmsb, xbz);
    { float* t = xa; xa = xbz; xbz = t; }
    gemm(hmsb, hmsb, HIDD, HIDD, oWt + (size_t)l * HIDD * HIDD, o_b + l * HIDD, 0, tmp, nullptr);
    add_kernel<<<NN * HIDD / 256, 256, 0, stream>>>(x, tmp, xb);
  }

  // out = x @ cls_W + cls_b
  head2_kernel<<<NN / 8, 256, 0, stream>>>(x, cls_W, cls_b, (float*)d_out, 0);
}
